// QuantLinear_70927089926463
// MI455X (gfx1250) — matrix operand order repaired and checked by reading, not compiled, not run on hardware
//
#include <hip/hip_runtime.h>

typedef _Float16 v16h __attribute__((ext_vector_type(16)));
typedef _Float16 v8h  __attribute__((ext_vector_type(8)));
typedef float    v8f  __attribute__((ext_vector_type(8)));
typedef int      v4i  __attribute__((ext_vector_type(4)));
typedef unsigned u8x  __attribute__((ext_vector_type(8)));

#define I_DIM  4096
#define O_DIM  11008
#define GROUP  128
#define NGROUP 32                          // I / GROUP
#define SPLITK 4
#define GRP_PER_SPLIT (NGROUP / SPLITK)    // 8 groups per wave
#define TILES_PER_BLOCK 2
#define TILES  (O_DIM / 16)                // 688
#define BLOCKS (TILES / TILES_PER_BLOCK)   // 344 blocks x 8 waves = 2752 waves

// {f16(1024 + lo.byte0), f16(1024 + hi.byte0)} in one dword:
// v_perm_b32 places byte0 of each source into the two half-slots (zeros elsewhere),
// v_or_b32 with 0x64006400 sets the fp16 exponent so value == 1024 + w exactly.
static __device__ __forceinline__ unsigned pack2(int lo, int hi) {
    return __builtin_amdgcn_perm((unsigned)hi, (unsigned)lo, 0x0c040c00u) | 0x64006400u;
}

__global__ __launch_bounds__(256)
void qlinear_wmma_splitk(const _Float16* __restrict__ x,
                         const int*      __restrict__ wq,
                         const _Float16* __restrict__ scale,
                         const _Float16* __restrict__ zero,
                         const _Float16* __restrict__ bias,
                         _Float16*       __restrict__ out) {
    __shared__ float red[8 * 32 * 8];   // [wave][lane][8] f32 partial accumulators

    const int tid  = threadIdx.x;
    const int wave = tid >> 5;
    const int lane = tid & 31;
    const int nl   = lane & 15;   // B/C column (= output channel), A row
    const int kh   = lane >> 4;   // K-half selector for A/B fragments
    const int tIn  = wave >> 2;   // tile within block (0..1)
    const int ks   = wave & 3;    // K-split index (0..3)

    const int tile = blockIdx.x * TILES_PER_BLOCK + tIn;
    const int o    = tile * 16 + nl;            // this lane's output channel

    const int*      wrow = wq + (size_t)o  * I_DIM;
    const _Float16* xrow = x  + (size_t)nl * I_DIM;

    v8f c = {};

    for (int gi = 0; gi < GRP_PER_SPLIT; ++gi) {
        const int g = ks * GRP_PER_SPLIT + gi;

        // group dequant constants: b_deq = v*s - m, v = 1024 + w
        const float sf = (float)scale[(size_t)o * NGROUP + g];
        const float zf = (float)zero [(size_t)o * NGROUP + g];
        const _Float16 s = (_Float16)sf;
        const _Float16 m = (_Float16)(sf * (1024.0f + zf));

        if (gi + 1 < GRP_PER_SPLIT)
            __builtin_prefetch(wrow + (g + 1) * GROUP + kh * 16, 0, 1);

        #pragma unroll
        for (int kk = 0; kk < 4; ++kk) {
            const int kbase = g * GROUP + kk * 32;

            // ---- B fragment: 16 consecutive uint8-in-int32 weights, NT streamed ----
            const int* wp = wrow + kbase + kh * 16;
            v4i w0 = __builtin_nontemporal_load((const v4i*)(wp + 0));
            v4i w1 = __builtin_nontemporal_load((const v4i*)(wp + 4));
            v4i w2 = __builtin_nontemporal_load((const v4i*)(wp + 8));
            v4i w3 = __builtin_nontemporal_load((const v4i*)(wp + 12));

            u8x p;
            p[0] = pack2(w0[0], w0[1]);
            p[1] = pack2(w0[2], w0[3]);
            p[2] = pack2(w1[0], w1[1]);
            p[3] = pack2(w1[2], w1[3]);
            p[4] = pack2(w2[0], w2[1]);
            p[5] = pack2(w2[2], w2[3]);
            p[6] = pack2(w3[0], w3[1]);
            p[7] = pack2(w3[2], w3[3]);
            v16h b = __builtin_bit_cast(v16h, p);
            b = b * s - m;                 // 8x v_pk_fma_f16

            // ---- A fragment (contiguous layout: 16 consecutive k at 16*kh) ----
            const _Float16* xp = xrow + kbase + kh * 16;
            v8h a0 = *(const v8h*)(xp);
            v8h a1 = *(const v8h*)(xp + 8);
            v16h a = __builtin_shufflevector(a0, a1,
                     0, 1, 2, 3, 4, 5, 6, 7, 8, 9, 10, 11, 12, 13, 14, 15);

            c = __builtin_amdgcn_wmma_f32_16x16x32_f16(
                    /*neg_a=*/false, a, /*neg_b=*/false, b,
                    /*c_mod=*/(short)0, c, /*reuse_a=*/false, /*reuse_b=*/false);
        }
    }

    // ---- split-K reduction through LDS ----
    *(v8f*)(red + ((wave * 32 + lane) << 3)) = c;
    __syncthreads();

    if (ks == 0) {   // wave-uniform branch, all WMMA already done
        v8f t = c;
        #pragma unroll
        for (int w2 = 1; w2 < SPLITK; ++w2)
            t += *(const v8f*)(red + (((wave + w2) * 32 + lane) << 3));

        const float bf = (float)bias[o];
        #pragma unroll
        for (int r = 0; r < 8; ++r)
            out[(size_t)(r + 8 * kh) * O_DIM + o] = (_Float16)(t[r] + bf);
    }
}

extern "C" void kernel_launch(void* const* d_in, const int* in_sizes, int n_in,
                              void* d_out, int out_size, void* d_ws, size_t ws_size,
                              hipStream_t stream) {
    (void)in_sizes; (void)n_in; (void)out_size; (void)d_ws; (void)ws_size;
    const _Float16* x     = (const _Float16*)d_in[0];
    const int*      wqp   = (const int*)     d_in[1];
    const _Float16* scale = (const _Float16*)d_in[2];
    const _Float16* zero  = (const _Float16*)d_in[3];
    const _Float16* bias  = (const _Float16*)d_in[4];
    _Float16*       out   = (_Float16*)d_out;

    qlinear_wmma_splitk<<<dim3(BLOCKS), dim3(256), 0, stream>>>(
        x, wqp, scale, zero, bias, out);
}
